// GRUNet_47364899340363
// MI455X (gfx1250) — compile-verified, matrix operand order checked by reading, not run on hardware
//
#include <hip/hip_runtime.h>
#include <hip/hip_bf16.h>

// ---------------------------------------------------------------------------
// GRUNet on gfx1250: f16 WMMA (f32 accum) + async-to-LDS weight staging.
// 64-wide K stages (2 WMMA slices) per barrier round; B fragments
// register-double-buffered so ds_load latency hides under WMMA execution.
// ---------------------------------------------------------------------------

typedef __attribute__((ext_vector_type(16))) _Float16 v16h;
typedef __attribute__((ext_vector_type(8)))  _Float16 v8h;
typedef __attribute__((ext_vector_type(8)))  float    v8f;

static constexpr int B  = 4096;
static constexpr int T  = 12;
static constexpr int I  = 256;
static constexpr int H  = 1024;
static constexpr int G3 = 3 * H;        // 3072
static constexpr int M  = B * T;        // 49152

// LDS tile pitch: 64 halfs of K + 8 halfs pad = 144B rows; 16B-aligned
// ds_load_b128 chunks, 36-bank row stride -> conflict-free 16-lane pattern.
static constexpr int LP = 72;

union FragU { v16h v; struct { v8h lo; v8h hi; } s; };

// Direct-from-global 16x32 f16 fragment (A operand).
// Per-lane layout (ISA 7.12.2, 16-bit A 16x32): lane L holds row (L&15),
// halfs [kb,kb+8) and [kb+16,kb+24), kb = (L>>4)*8.
__device__ __forceinline__ v16h load_frag16(const _Float16* __restrict__ base,
                                            long ld, int lane) {
  const int r  = lane & 15;
  const int kb = (lane >> 4) << 3;   // 0 or 8
  const _Float16* p = base + (long)r * ld + kb;
  FragU u;
  u.s.lo = *(const v8h*)(p);
  u.s.hi = *(const v8h*)(p + 16);
  return u.v;
}

// Fragment from an LDS-staged tile (pitch LP halfs).
// row0 = first N-row of the 16x16 tile; kofs = 0 or 32 (slice within stage).
__device__ __forceinline__ v16h lds_frag16(const _Float16* buf, int row0,
                                           int kofs, int lane) {
  const int r  = row0 + (lane & 15);
  const int kb = (lane >> 4) << 3;
  const _Float16* p = buf + r * LP + kofs + kb;
  FragU u;
  u.s.lo = *(const v8h*)(p);        // ds_load_b128
  u.s.hi = *(const v8h*)(p + 16);   // ds_load_b128
  return u.v;
}

__device__ __forceinline__ v8f wmma_f16(v16h a, v16h b, v8f c) {
  return __builtin_amdgcn_wmma_f32_16x16x32_f16(false, a, false, b,
                                                (short)0, c, false, false);
}

// One per-lane 16B global->LDS async copy (ASYNCcnt-tracked, no VGPR data).
__device__ __forceinline__ void async_copy16(const _Float16* gp, _Float16* lp) {
  unsigned lds_off = (unsigned)(unsigned long long)(uintptr_t)lp;
  unsigned long long gaddr = (unsigned long long)(uintptr_t)gp;
  asm volatile("global_load_async_to_lds_b128 %0, %1, off"
               :: "v"(lds_off), "v"(gaddr) : "memory");
}

__device__ __forceinline__ void wait_async0() {
  asm volatile("s_wait_asynccnt 0x0" ::: "memory");
}

__device__ __forceinline__ float sigmoidf_fast(float x) {
  return 1.0f / (1.0f + __expf(-x));
}

// ---------------------------------------------------------------------------
// f32 -> f16 conversion / zero fill
// ---------------------------------------------------------------------------
__global__ void cvt_f32_to_f16(const float* __restrict__ src,
                               _Float16* __restrict__ dst, int n) {
  int i = blockIdx.x * blockDim.x + threadIdx.x;
  if (i < n) dst[i] = (_Float16)src[i];
}

__global__ void fill_zero_f16(_Float16* __restrict__ dst, int n) {
  int i = blockIdx.x * blockDim.x + threadIdx.x;
  if (i < n) dst[i] = (_Float16)0.0f;
}

// ---------------------------------------------------------------------------
// Batched input GEMM: out[m][n] = sum_k A[m][k]*W[n][k] + bias[n]  (f16 out)
// Block = 256 thr (8 waves); block tile 128(M) x 64(N); wave tile 16x64.
// Weight stage: 64 rows x 64 halfs in LDS, double-buffered, async-copied.
// ---------------------------------------------------------------------------
__global__ void __launch_bounds__(256)
gemm_xi_kernel(const _Float16* __restrict__ A, int K,
               const _Float16* __restrict__ W,
               const float* __restrict__ bias,
               _Float16* __restrict__ out) {
  __shared__ __align__(16) _Float16 Wsh[2][64 * LP];   // 2 x 9216B

  const int tid   = threadIdx.x;
  const int lane  = tid & 31;
  const int wave  = tid >> 5;
  const int mbase = blockIdx.x * 128 + wave * 16;
  const int nbase = blockIdx.y * 64;

  // Stage one 64x64-half weight tile: 512 16B chunks, 2 per thread.
  auto kick = [&](int bufi, int kk) {
#pragma unroll
    for (int j = 0; j < 2; ++j) {
      const int c   = j * 256 + tid;     // 0..511
      const int row = c >> 3;            // 0..63 (N row within tile)
      const int q   = (c & 7) << 3;      // 0..56 halfs along K
      const _Float16* gp = W + (long)(nbase + row) * K + kk + q;
      async_copy16(gp, &Wsh[bufi][row * LP + q]);
    }
  };

  v8f acc[4];
#pragma unroll
  for (int s = 0; s < 4; ++s) acc[s] = (v8f)0.0f;

  const _Float16* arow = A + (long)mbase * K;

  kick(0, 0);
  for (int kk = 0; kk < K; kk += 64) {
    const int cur = (kk >> 6) & 1;
    // A loads are independent of the LDS staging: issue before the waits so
    // their global latency overlaps the async wait + barrier.
    v16h a0 = load_frag16(arow + kk,      K, lane);
    v16h a1 = load_frag16(arow + kk + 32, K, lane);

    wait_async0();        // own copies for this stage done
    __syncthreads();      // all copies landed + previous compute finished
    if (kk + 64 < K) kick(cur ^ 1, kk + 64);

    // 8 WMMAs, B fragment register-double-buffered: load frag i+1 while
    // WMMA i executes (order: s-major, slice-minor).
    v16h bf = lds_frag16(&Wsh[cur][0], 0, 0, lane);
#pragma unroll
    for (int i = 0; i < 8; ++i) {
      v16h bc = bf;
      if (i + 1 < 8) {
        const int i1 = i + 1;
        bf = lds_frag16(&Wsh[cur][0], (i1 >> 1) * 16, (i1 & 1) * 32, lane);
      }
      acc[i >> 1] = wmma_f16((i & 1) ? a1 : a0, bc, acc[i >> 1]);
    }
  }

  // Epilogue: D layout — VGPR i: row = mbase + i + 8*(lane>>4), col = lane&15.
  const int col    = lane & 15;
  const int rowoff = (lane >> 4) * 8;
#pragma unroll
  for (int s = 0; s < 4; ++s) {
    const int ncol = nbase + s * 16 + col;
    const float bn = bias[ncol];
#pragma unroll
    for (int i = 0; i < 8; ++i) {
      const int row = mbase + i + rowoff;
      out[(long)row * G3 + ncol] = (_Float16)(acc[s][i] + bn);
    }
  }
}

// ---------------------------------------------------------------------------
// Fused GRU step: gh = h_prev @ W_hh^T (+b_hh) for all 3 gates of a 16x64
// column tile via WMMA, combine with xi, apply GRU nonlinearity, write h_t.
// Grid: (B/128, H/64), block = 256 thr (8 waves).
// Weight stage: 3 gates x 64 rows x 64 halfs in LDS, double-buffered.
// ---------------------------------------------------------------------------
__global__ void __launch_bounds__(256)
gru_step_kernel(const _Float16* __restrict__ hprev, long ldh_prev,
                const _Float16* __restrict__ Whh,          // [3H][H] f16
                const float* __restrict__ bhh,             // [3H]
                const _Float16* __restrict__ xi_t, long ldxi,
                _Float16* __restrict__ hout, long ldh_out) {
  __shared__ __align__(16) _Float16 Bsh[2][192 * LP];   // 2 x 27648B

  const int tid   = threadIdx.x;
  const int lane  = tid & 31;
  const int wave  = tid >> 5;
  const int mbase = blockIdx.x * 128 + wave * 16;
  const int nbase = blockIdx.y * 64;

  // Stage 192 rows x 64 halfs: 1536 16B chunks, 6 per thread.
  auto kick = [&](int bufi, int kk) {
#pragma unroll
    for (int j = 0; j < 6; ++j) {
      const int c   = j * 256 + tid;     // 0..1535
      const int row = c >> 3;            // 0..191 = g*64 + n_local
      const int q   = (c & 7) << 3;      // 0..56 halfs
      const int g   = row >> 6;
      const int nl  = row & 63;
      const _Float16* gp = Whh + (long)(g * H + nbase + nl) * H + kk + q;
      async_copy16(gp, &Bsh[bufi][row * LP + q]);
    }
  };

  v8f acc[3][4];
#pragma unroll
  for (int g = 0; g < 3; ++g)
#pragma unroll
    for (int s = 0; s < 4; ++s) acc[g][s] = (v8f)0.0f;

  const _Float16* arow = hprev + (long)mbase * ldh_prev;

  kick(0, 0);
  for (int kk = 0; kk < H; kk += 64) {
    const int cur = (kk >> 6) & 1;
    // Hoist A loads above the waits (independent of LDS staging).
    v16h a0 = load_frag16(arow + kk,      ldh_prev, lane);
    v16h a1 = load_frag16(arow + kk + 32, ldh_prev, lane);

    wait_async0();        // own copies for this stage done
    __syncthreads();      // all copies landed + previous compute finished
    if (kk + 64 < H) kick(cur ^ 1, kk + 64);

    // 24 WMMAs, B fragment register-double-buffered.
    // Index order: g-major, s, slice-minor: i = g*8 + s*2 + slice.
    v16h bf = lds_frag16(&Bsh[cur][0], 0, 0, lane);
#pragma unroll
    for (int i = 0; i < 24; ++i) {
      v16h bc = bf;
      if (i + 1 < 24) {
        const int i1 = i + 1;
        const int r1 = (i1 >> 3) * 64 + ((i1 >> 1) & 3) * 16;
        bf = lds_frag16(&Bsh[cur][0], r1, (i1 & 1) * 32, lane);
      }
      acc[i >> 3][(i >> 1) & 3] =
          wmma_f16((i & 1) ? a1 : a0, bc, acc[i >> 3][(i >> 1) & 3]);
    }
  }

  // Fused GRU gate epilogue (PyTorch gate order r, z, n):
  //   r = sig(ir + hr); z = sig(iz + hz); n = tanh(in + r*hn)
  //   h' = (1-z)*n + z*h
  const int col    = lane & 15;
  const int rowoff = (lane >> 4) * 8;
#pragma unroll
  for (int s = 0; s < 4; ++s) {
    const int ncol = nbase + s * 16 + col;   // column within H
    const float br = bhh[0 * H + ncol];
    const float bz = bhh[1 * H + ncol];
    const float bn = bhh[2 * H + ncol];
#pragma unroll
    for (int i = 0; i < 8; ++i) {
      const int row = mbase + i + rowoff;
      const long xoff = (long)row * ldxi;
      const float ir  = (float)xi_t[xoff + 0 * H + ncol];
      const float iz  = (float)xi_t[xoff + 1 * H + ncol];
      const float inn = (float)xi_t[xoff + 2 * H + ncol];
      const float hp  = (float)hprev[(long)row * ldh_prev + ncol];
      const float hr  = acc[0][s][i] + br;
      const float hz  = acc[1][s][i] + bz;
      const float hn  = acc[2][s][i] + bn;
      const float r = sigmoidf_fast(ir + hr);
      const float z = sigmoidf_fast(iz + hz);
      const float n = tanhf(inn + r * hn);
      const float hnew = (1.0f - z) * n + z * hp;
      hout[(long)row * ldh_out + ncol] = (_Float16)hnew;
    }
  }
}

// ---------------------------------------------------------------------------
// Head: out[m] = relu(h2[m][:]) . W_lin + b_lin     (one wave per row)
// ---------------------------------------------------------------------------
__global__ void __launch_bounds__(256)
head_kernel(const _Float16* __restrict__ h2,
            const float* __restrict__ wlin,
            const float* __restrict__ blin,
            float* __restrict__ out) {
  const int lane = threadIdx.x & 31;
  const int wave = threadIdx.x >> 5;
  const int row  = blockIdx.x * 8 + wave;

  const _Float16* hr = h2 + (long)row * H;
  float sum = 0.0f;
  for (int k = lane; k < H; k += 32) {
    float v = (float)hr[k];
    v = fmaxf(v, 0.0f);
    sum += v * wlin[k];
  }
#pragma unroll
  for (int off = 16; off > 0; off >>= 1)
    sum += __shfl_xor(sum, off, 32);
  if (lane == 0) out[row] = sum + blin[0];
}

// ---------------------------------------------------------------------------
// Launch
// ---------------------------------------------------------------------------
extern "C" void kernel_launch(void* const* d_in, const int* in_sizes, int n_in,
                              void* d_out, int out_size, void* d_ws, size_t ws_size,
                              hipStream_t stream) {
  const float* x     = (const float*)d_in[0];
  const float* Wih0  = (const float*)d_in[1];
  const float* Whh0  = (const float*)d_in[2];
  const float* bih0  = (const float*)d_in[3];
  const float* bhh0  = (const float*)d_in[4];
  const float* Wih1  = (const float*)d_in[5];
  const float* Whh1  = (const float*)d_in[6];
  const float* bih1  = (const float*)d_in[7];
  const float* bhh1  = (const float*)d_in[8];
  const float* Wlin  = (const float*)d_in[9];
  const float* blin  = (const float*)d_in[10];
  float* out = (float*)d_out;

  // Workspace carve (256B aligned)
  char* p = (char*)d_ws;
  auto carve = [&](size_t bytes) -> void* {
    void* r = (void*)p;
    p += (bytes + 255) & ~(size_t)255;
    return r;
  };
  _Float16* x16     = (_Float16*)carve((size_t)M  * I  * 2);  //  25 MB
  _Float16* wih0_16 = (_Float16*)carve((size_t)G3 * I  * 2);  // 1.5 MB
  _Float16* whh0_16 = (_Float16*)carve((size_t)G3 * H  * 2);  //   6 MB
  _Float16* wih1_16 = (_Float16*)carve((size_t)G3 * H  * 2);  //   6 MB
  _Float16* whh1_16 = (_Float16*)carve((size_t)G3 * H  * 2);  //   6 MB
  _Float16* xi16    = (_Float16*)carve((size_t)M  * G3 * 2);  // 302 MB (reused)
  _Float16* h1      = (_Float16*)carve((size_t)M  * H  * 2);  // 100 MB [B,T,H]
  _Float16* h2      = (_Float16*)carve((size_t)M  * H  * 2);  // 100 MB [B,T,H]
  _Float16* hzero   = (_Float16*)carve((size_t)B  * H  * 2);  //   8 MB
  (void)ws_size; (void)n_in; (void)in_sizes; (void)out_size;

  auto cvt = [&](const float* s, _Float16* d, int n) {
    cvt_f32_to_f16<<<(n + 255) / 256, 256, 0, stream>>>(s, d, n);
  };
  cvt(x,    x16,     M * I);
  cvt(Wih0, wih0_16, G3 * I);
  cvt(Whh0, whh0_16, G3 * H);
  cvt(Wih1, wih1_16, G3 * H);
  cvt(Whh1, whh1_16, G3 * H);
  {
    int n = B * H;
    fill_zero_f16<<<(n + 255) / 256, 256, 0, stream>>>(hzero, n);
  }

  const dim3 gemm_grid(M / 128, G3 / 64);   // 384 x 48
  const dim3 step_grid(B / 128, H / 64);    //  32 x 16

  // ---- Layer 0 ----
  gemm_xi_kernel<<<gemm_grid, 256, 0, stream>>>(x16, I, wih0_16, bih0, xi16);
  for (int t = 0; t < T; ++t) {
    const _Float16* hp = (t == 0) ? hzero : (h1 + (long)(t - 1) * H);
    const long ldp     = (t == 0) ? (long)H : (long)T * H;
    gru_step_kernel<<<step_grid, 256, 0, stream>>>(
        hp, ldp, whh0_16, bhh0,
        xi16 + (long)t * G3, (long)T * G3,
        h1 + (long)t * H, (long)T * H);
  }

  // ---- Layer 1 ---- (h1 is [B,T,H] flat == [M][H] row-major GEMM input)
  gemm_xi_kernel<<<gemm_grid, 256, 0, stream>>>(h1, H, wih1_16, bih1, xi16);
  for (int t = 0; t < T; ++t) {
    const _Float16* hp = (t == 0) ? hzero : (h2 + (long)(t - 1) * H);
    const long ldp     = (t == 0) ? (long)H : (long)T * H;
    gru_step_kernel<<<step_grid, 256, 0, stream>>>(
        hp, ldp, whh1_16, bhh1,
        xi16 + (long)t * G3, (long)T * G3,
        h2 + (long)t * H, (long)T * H);
  }

  // ---- Head ----
  head_kernel<<<M / 8, 256, 0, stream>>>(h2, Wlin, blin, out);
}
